// CriticNetwork_3994319585784
// MI455X (gfx1250) — matrix operand order repaired and checked by reading, not compiled, not run on hardware
//
#include <hip/hip_runtime.h>
#include <hip/hip_bf16.h>
#include <stdint.h>
#include <stddef.h>

// MI455X (gfx1250) implementation.
// Precision strategy: f16 WMMA (V_WMMA_F32_16X16X32_F16) with fp32 accumulation.
// HBM floor ~160MB => ~7us @23.3TB/s; x (92MB) stays L2-resident (192MB L2),
// so the sequential GRU time loop re-reads from L2, not HBM.
// B operands are staged in LDS in *fragment-linear* order: each lane reads its
// 32B of operand with 2x ds_load_b128, no register repacking.
// LM2's A tile is staged with gfx1250 GLOBAL_LOAD_ASYNC_TO_LDS_B128.

typedef _Float16 half_t;
typedef __attribute__((ext_vector_type(16))) _Float16 v16h;
typedef __attribute__((ext_vector_type(8)))  _Float16 v8h;
typedef __attribute__((ext_vector_type(2)))  _Float16 half2_t;
typedef __attribute__((ext_vector_type(8)))  float    v8f;
typedef int vec4i __attribute__((__vector_size__(16)));

#define DEVINL __device__ __forceinline__

static constexpr int kN = 100;     // nodes
static constexpr int kB = 128;     // batch
static constexpr int kT = 256;     // timesteps
static constexpr int kS = 7;       // raw features per node
static constexpr int kM = kN * kS; // 700
static constexpr int kF = 35;      // gathered features (5 neighbors * 7)
static constexpr int kG = 16;      // GRU hidden

static constexpr size_t XT2_BYTES = (size_t)kT * kM * kB * sizeof(half_t);   // 45,875,200
static constexpr size_t H1_BYTES  = (size_t)kN * kB * 512 * sizeof(half_t);  // 13,107,200

// ---- gfx1250 async global->LDS path (GLOBAL_LOAD_ASYNC_TO_LDS_B128) --------
#if defined(__has_builtin)
#  if __has_builtin(__builtin_amdgcn_global_load_async_to_lds_b128)
#    define USE_ASYNC_COPY 1
#  endif
#endif
#ifndef USE_ASYNC_COPY
#  define USE_ASYNC_COPY 0
#endif

DEVINL void async_copy16(const half_t* g, half_t* l) {
#if USE_ASYNC_COPY
  __builtin_amdgcn_global_load_async_to_lds_b128(
      (__attribute__((address_space(1))) vec4i*)g,
      (__attribute__((address_space(3))) vec4i*)l, 0, 0);
#else
  *(v8h*)l = *(const v8h*)g;
#endif
}
DEVINL void async_join() {
#if USE_ASYNC_COPY
#  if __has_builtin(__builtin_amdgcn_s_wait_asynccnt)
  __builtin_amdgcn_s_wait_asynccnt(0);
#  else
  asm volatile("s_wait_asynccnt 0" ::: "memory");
#  endif
#endif
}

DEVINL v8f wmma16(v16h a, v16h b, v8f c) {
  // D = A(16x32 f16) x B(32x16 f16) + C(f32)
  return __builtin_amdgcn_wmma_f32_16x16x32_f16(false, a, false, b, (short)0, c, false, false);
}

DEVINL v8f zf8() {
  v8f z;
#pragma unroll
  for (int i = 0; i < 8; ++i) z[i] = 0.f;
  return z;
}

DEVINL v16h cat16(v8h lo, v8h hi8) {
  v16h r;
#pragma unroll
  for (int i = 0; i < 8; ++i) { r[i] = lo[i]; r[8 + i] = hi8[i]; }
  return r;
}

// A-fragment (16x32, row-major source, stride in halves).
// Lane l<16: row=base+l, K 0..15; lane l>=16: K 16..31.
DEVINL v16h afrag(const half_t* base, int strideH, int row, int kb, int hi) {
  const half_t* p = base + row * strideH + kb + hi * 16;
  v8h lo = *(const v8h*)p;
  v8h hh = *(const v8h*)(p + 8);
  return cat16(lo, hh);
}
// A-fragment with only K=0..15 valid (upper 16 K zero).
DEVINL v16h afrag_k16(const half_t* base, int strideH, int row, int cb, int hi) {
  const half_t* p = base + row * strideH + cb;
  v8h lo = *(const v8h*)p;
  v8h hh = *(const v8h*)(p + 8);
  v16h r;
#pragma unroll
  for (int i = 0; i < 8; ++i) { r[i] = hi ? (half_t)0.f : lo[i]; r[8 + i] = hi ? (half_t)0.f : hh[i]; }
  return r;
}

// ---- fragment-linear B operand staging --------------------------------------
// Element (kLocal in 0..31, col in 0..15) of fragment `frag` lives at:
DEVINL int fragidx(int frag, int kLocal, int col) {
  return ((frag * 32) + ((kLocal >> 4) * 16) + col) * 16 + (kLocal & 15);
}
// Load = 32 contiguous bytes per lane -> two ds_load_b128, zero repacking.
DEVINL v16h bfrag_lin(const half_t* Wf, int frag, int lane) {
  const half_t* p = Wf + ((size_t)(frag * 32 + lane)) * 16;
  v8h lo = *(const v8h*)p;
  v8h hh = *(const v8h*)(p + 8);
  return cat16(lo, hh);
}
// B-fragment from "kpair" LDS layout where only K rows 0..15 exist (cold path).
DEVINL v16h bfrag_k16(const half_t* W, int nc2, int col, int hi) {
  v16h r;
#pragma unroll
  for (int q = 0; q < 16; ++q) r[q] = (half_t)0.f;
  if (hi == 0) {
    const half_t* p = W + col * 2;
#pragma unroll
    for (int q = 0; q < 8; ++q) {
      half2_t v = *(const half2_t*)(p + q * nc2);
      r[2 * q] = v[0]; r[2 * q + 1] = v[1];
    }
  }
  return r;
}

DEVINL float sigm(float x) { return 1.0f / (1.0f + __expf(-x)); }
DEVINL float tanhfast(float x) {
  x = fminf(fmaxf(x, -15.f), 15.f);
  float e = __expf(-2.0f * x);
  return (1.0f - e) / (1.0f + e);
}

// ---------------------------------------------------------------------------
// K0: x (B,T,N,S) f32  ->  xT2 (T, N*S, B) f16   (b-major, coalesced for GRU)
// ---------------------------------------------------------------------------
__global__ __launch_bounds__(256) void k_transpose(const float* __restrict__ x,
                                                   half_t* __restrict__ xT2) {
  __shared__ float tile[32][33];
  const int t  = blockIdx.x;
  const int m0 = blockIdx.y * 32;
  const int b0 = blockIdx.z * 32;
  const int tx = threadIdx.x & 31;
  const int ty = threadIdx.x >> 5;
#pragma unroll
  for (int i = 0; i < 4; ++i) {
    int m = m0 + tx;
    int bb = b0 + ty + i * 8;
    float v = 0.f;
    if (m < kM) v = x[((size_t)bb * kT + t) * kM + m];
    tile[ty + i * 8][tx] = v;
  }
  __syncthreads();
#pragma unroll
  for (int i = 0; i < 4; ++i) {
    int m = m0 + ty + i * 8;
    int bb = b0 + tx;
    if (m < kM) xT2[((size_t)t * kM + m) * kB + bb] = (half_t)tile[tx][ty + i * 8];
  }
}

// ---------------------------------------------------------------------------
// LM1: H1 = relu(Xlast(128x35) @ W1(35x512) + b1) per node.  grid (N, 8).
// ---------------------------------------------------------------------------
__global__ __launch_bounds__(256) void k_lm1(const half_t* __restrict__ xT2,
                                             const float* __restrict__ W1,
                                             const float* __restrict__ b1,
                                             half_t* __restrict__ H1) {
  __shared__ alignas(16) half_t xs[128 * 64];      // gathered features, K padded to 64
  __shared__ alignas(16) half_t Wf[8 * 32 * 16];   // 8 fragment-linear B frags
  __shared__ float b1s[64];
  const int n = blockIdx.x;
  const int c0 = blockIdx.y * 64;
  const int tid = threadIdx.x;
  const int lane = tid & 31, wave = tid >> 5;
  const int lh = lane & 15, hi = lane >> 4;
  const int mb = wave * 16;

  int nbase[5], nvalid[5];
  {
    const int offs[5] = {0, -10, 1, 10, -1};
#pragma unroll
    for (int k = 0; k < 5; ++k) {
      int nn = n + offs[k];
      nvalid[k] = (nn >= 0 && nn < kN);
      nn = nn < 0 ? 0 : (nn >= kN ? kN - 1 : nn);
      nbase[k] = nn * kS;
    }
  }
  {
    uint32_t* xz = (uint32_t*)xs;
    for (int i = tid; i < 128 * 32; i += 256) xz[i] = 0u;
  }
  __syncthreads();
  // gather features at t = T-1
#pragma unroll
  for (int k = 0; k < 5; ++k) {
#pragma unroll
    for (int j = 0; j < 2; ++j) {
      int e = tid + j * 256;
      if (e < 448) {
        int s = e >> 6, db = e & 63;
        half2_t v; v[0] = (half_t)0.f; v[1] = (half_t)0.f;
        if (nvalid[k])
          v = *(const half2_t*)(xT2 + ((size_t)(kT - 1) * kM + nbase[k] + s) * kB + db * 2);
        xs[(db * 2) * 64 + k * kS + s] = v[0];
        xs[(db * 2 + 1) * 64 + k * kS + s] = v[1];
      }
    }
  }
  // stage W1 tile (rows 0..34 valid, rest zero), fragment-linear
  for (int i = tid; i < 64 * 64; i += 256) {
    int k = i >> 6, c = i & 63;
    float v = (k < kF) ? W1[((size_t)n * kF + k) * 512 + c0 + c] : 0.f;
    int frag = (c >> 4) * 2 + (k >> 5);
    Wf[fragidx(frag, k & 31, c & 15)] = (half_t)v;
  }
  if (tid < 64) b1s[tid] = b1[(size_t)n * 512 + c0 + tid];
  __syncthreads();

  v16h a0 = afrag(xs, 64, mb + lh, 0, hi);
  v16h a1 = afrag(xs, 64, mb + lh, 32, hi);
#pragma unroll
  for (int j = 0; j < 4; ++j) {
    v16h w0 = bfrag_lin(Wf, j * 2 + 0, lane);
    v16h w1 = bfrag_lin(Wf, j * 2 + 1, lane);
    v8f c = zf8();
    c = wmma16(a0, w0, c);
    c = wmma16(a1, w1, c);
    float bias = b1s[j * 16 + lh];
#pragma unroll
    for (int v = 0; v < 8; ++v) {
      float y = c[v] + bias;
      y = y > 0.f ? y : 0.f;
      H1[((size_t)n * kB + mb + hi * 8 + v) * 512 + c0 + j * 16 + lh] = (half_t)y;
    }
  }
}

// ---------------------------------------------------------------------------
// LM2: H2 = relu(H1(128x512) @ W2(512x256) + b2) per node.  grid (N, 4).
// A tile staged via gfx1250 async global->LDS; W tile converted f32->f16 into
// fragment-linear layout.
// ---------------------------------------------------------------------------
__global__ __launch_bounds__(256) void k_lm2(const half_t* __restrict__ H1,
                                             const float* __restrict__ W2,
                                             const float* __restrict__ b2,
                                             half_t* __restrict__ H2) {
  __shared__ alignas(16) half_t As[128 * 32];
  __shared__ alignas(16) half_t Wf[4 * 32 * 16];   // 4 fragment-linear B frags
  __shared__ float b2s[64];
  const int n = blockIdx.x;
  const int c0 = blockIdx.y * 64;
  const int tid = threadIdx.x;
  const int lane = tid & 31, wave = tid >> 5;
  const int lh = lane & 15, hi = lane >> 4;
  const int mb = wave * 16;
  if (tid < 64) b2s[tid] = b2[(size_t)n * 256 + c0 + tid];

  v8f acc[4];
#pragma unroll
  for (int j = 0; j < 4; ++j) acc[j] = zf8();

  for (int k0 = 0; k0 < 512; k0 += 32) {
    if (tid < 32 && k0 + 32 < 512)  // gfx1250 global_prefetch_b8 for next W2 tile
      __builtin_prefetch(W2 + ((size_t)n * 512 + k0 + 32 + tid) * 256 + c0, 0, 0);
    __syncthreads();  // all waves done reading previous As/Wf
    // A tile: 512 x 16B layout-preserving chunks -> async LDS copy
#pragma unroll
    for (int u = 0; u < 2; ++u) {
      int ch = tid + u * 256;
      int row = ch >> 2, part = ch & 3;
      async_copy16(H1 + ((size_t)n * kB + row) * 512 + k0 + part * 8,
                   As + row * 32 + part * 8);
    }
    // W tile: f32 -> f16 with fragment-linear scatter
    for (int i = tid; i < 2048; i += 256) {
      int k = i >> 6, c = i & 63;
      Wf[fragidx(c >> 4, k, c & 15)] =
          (half_t)W2[((size_t)n * 512 + k0 + k) * 256 + c0 + c];
    }
    async_join();
    __syncthreads();
    v16h a = afrag(As, 32, mb + lh, 0, hi);
#pragma unroll
    for (int j = 0; j < 4; ++j)
      acc[j] = wmma16(a, bfrag_lin(Wf, j, lane), acc[j]);
  }
#pragma unroll
  for (int j = 0; j < 4; ++j) {
    float bias = b2s[j * 16 + lh];
#pragma unroll
    for (int v = 0; v < 8; ++v) {
      float y = acc[j][v] + bias;
      y = y > 0.f ? y : 0.f;
      H2[((size_t)n * kB + mb + hi * 8 + v) * 256 + c0 + j * 16 + lh] = (half_t)y;
    }
  }
}

// ---------------------------------------------------------------------------
// LM3: f[b] = H2[b,:] . W3 + b3 ; out[n*128+b] = f.  grid (N), 128 threads.
// ---------------------------------------------------------------------------
__global__ __launch_bounds__(128) void k_lm3(const half_t* __restrict__ H2,
                                             const float* __restrict__ W3,
                                             const float* __restrict__ b3,
                                             float* __restrict__ out) {
  __shared__ float w3s[256];
  const int n = blockIdx.x;
  const int tid = threadIdx.x;
  for (int i = tid; i < 256; i += 128) w3s[i] = W3[(size_t)n * 256 + i];
  __syncthreads();
  const half_t* row = H2 + ((size_t)n * kB + tid) * 256;
  float acc = b3[n];
#pragma unroll
  for (int i = 0; i < 256; i += 8) {
    v8h v = *(const v8h*)(row + i);
#pragma unroll
    for (int j = 0; j < 8; ++j) acc += (float)v[j] * w3s[i + j];
  }
  out[(size_t)n * kB + tid] = acc;
}

// ---------------------------------------------------------------------------
// GRU + score MLP.  grid (N, 2): one workgroup per (node, batch-half),
// 128 threads = 4 waves, each wave owns a 16-row batch tile.
// Fused operand: xs2 row b = [35 features | zeros | h(16)] (K=64),
// W operand     = [Wih^T rows 0..34 ; zeros ; Whh^T(r,z) rows 48..63].
// Extra 1 WMMA computes h@Whh_n for the r-gated candidate gate.
// ---------------------------------------------------------------------------
__global__ __launch_bounds__(128) void k_gru(const half_t* __restrict__ xT2,
                                             const float* __restrict__ Wih,
                                             const float* __restrict__ Whh,
                                             const float* __restrict__ bih,
                                             const float* __restrict__ bhh,
                                             const float* __restrict__ sW1,
                                             const float* __restrict__ sb1g,
                                             const float* __restrict__ sW2,
                                             const float* __restrict__ sb2g,
                                             const float* __restrict__ sW3,
                                             const float* __restrict__ sb3g,
                                             float* __restrict__ out) {
  __shared__ alignas(16) half_t xs[2][64 * 64];    // double-buffered [x|pad|h]
  __shared__ alignas(16) half_t Wf[6 * 32 * 16];   // 6 fragment-linear B frags
  __shared__ alignas(16) half_t sW1l[8 * 32];      // kpair, K=16 x 16
  __shared__ alignas(16) half_t sW2l[8 * 32];
  __shared__ alignas(16) half_t gs[64 * 16];
  __shared__ float bihs[48], bhhs[48], sb1s[16], sb2s[16], sw3s[16];

  const int n = blockIdx.x;
  const int yb = blockIdx.y;          // batch half
  const int tid = threadIdx.x;
  const int lane = tid & 31, wave = tid >> 5;
  const int lh = lane & 15, hi = lane >> 4;
  const int mb = wave * 16;           // local row base

  int nbase[5], nvalid[5];
  {
    const int offs[5] = {0, -10, 1, 10, -1};
#pragma unroll
    for (int k = 0; k < 5; ++k) {
      int nn = n + offs[k];
      nvalid[k] = (nn >= 0 && nn < kN);
      nn = nn < 0 ? 0 : (nn >= kN ? kN - 1 : nn);
      nbase[k] = nn * kS;
    }
  }
  {
    uint32_t* xz = (uint32_t*)&xs[0][0];
    for (int i = tid; i < 2 * 64 * 32; i += 128) xz[i] = 0u;  // pads + h0 = 0
  }
  // combined [Wih^T ; 0 ; Whh^T(r,z)] operand, fragment-linear
  for (int i = tid; i < 64 * 48; i += 128) {
    int k = i / 48, c = i % 48;
    float v = 0.f;
    if (k < kF) v = Wih[((size_t)n * 48 + c) * kF + k];
    else if (k >= 48 && c < 32) v = Whh[((size_t)n * 48 + c) * kG + (k - 48)];
    int frag = (c >> 4) * 2 + (k >> 5);
    Wf[fragidx(frag, k & 31, c & 15)] = (half_t)v;
  }
  for (int i = tid; i < 48; i += 128) {
    bihs[i] = bih[(size_t)n * 48 + i];
    bhhs[i] = bhh[(size_t)n * 48 + i];
  }
  for (int i = tid; i < 256; i += 128) {
    int k = i >> 4, c = i & 15;
    sW1l[(k >> 1) * 32 + c * 2 + (k & 1)] = (half_t)sW1[((size_t)n * kG + k) * kG + c];
    sW2l[(k >> 1) * 32 + c * 2 + (k & 1)] = (half_t)sW2[((size_t)n * kG + k) * kG + c];
  }
  if (tid < 16) {
    sb1s[tid] = sb1g[(size_t)n * kG + tid];
    sb2s[tid] = sb2g[(size_t)n * kG + tid];
    sw3s[tid] = sW3[(size_t)n * kG + tid];
  }
  __syncthreads();

  // preload t = 0 features into buffer 0
#pragma unroll
  for (int k = 0; k < 5; ++k) {
#pragma unroll
    for (int j = 0; j < 2; ++j) {
      int e = tid + j * 128;
      if (e < 224) {
        int s = e >> 5, db = e & 31;
        half2_t v; v[0] = (half_t)0.f; v[1] = (half_t)0.f;
        if (nvalid[k])
          v = *(const half2_t*)(xT2 + ((size_t)nbase[k] + s) * kB + yb * 64 + db * 2);
        xs[0][(db * 2) * 64 + k * kS + s] = v[0];
        xs[0][(db * 2 + 1) * 64 + k * kS + s] = v[1];
      }
    }
  }

  // persistent B-fragments (weights are time-invariant)
  v16h bf00 = bfrag_lin(Wf, 0, lane);
  v16h bf01 = bfrag_lin(Wf, 1, lane);
  v16h bf10 = bfrag_lin(Wf, 2, lane);
  v16h bf11 = bfrag_lin(Wf, 3, lane);
  v16h bf20 = bfrag_lin(Wf, 4, lane);
  v16h bf21 = bfrag_lin(Wf, 5, lane);
  v16h bfe;
  {
#pragma unroll
    for (int q = 0; q < 16; ++q) bfe[q] = (half_t)0.f;
    if (hi == 0) {
      const float* wp = Whh + ((size_t)n * 48 + 32 + lh) * kG;
#pragma unroll
      for (int q = 0; q < 16; ++q) bfe[q] = (half_t)wp[q];
    }
  }
  __syncthreads();

  const float brz0 = bihs[lh] + bhhs[lh];
  const float brz1 = bihs[16 + lh] + bhhs[16 + lh];
  const float bin  = bihs[32 + lh];
  const float bhn  = bhhs[32 + lh];

  v8f hreg = zf8();
  int cur = 0;
  for (int t = 0; t < kT; ++t) {
    const bool more = (t + 1 < kT);
    if (tid < 5 && t + 2 < kT) {
      int off = (tid == 0) ? 0 : (tid == 1) ? -10 : (tid == 2) ? 1 : (tid == 3) ? 10 : -1;
      int nn = n + off; nn = nn < 0 ? 0 : (nn >= kN ? kN - 1 : nn);
      __builtin_prefetch(xT2 + ((size_t)(t + 2) * kM + nn * kS) * kB + yb * 64, 0, 0);
    }
    // issue global loads for step t+1 early (hidden under WMMAs)
    half2_t fv[5][2];
#pragma unroll
    for (int k = 0; k < 5; ++k) {
#pragma unroll
      for (int j = 0; j < 2; ++j) {
        fv[k][j][0] = (half_t)0.f; fv[k][j][1] = (half_t)0.f;
        int e = tid + j * 128;
        if (more && e < 224 && nvalid[k]) {
          int s = e >> 5, db = e & 31;
          fv[k][j] = *(const half2_t*)(xT2 + ((size_t)(t + 1) * kM + nbase[k] + s) * kB +
                                       yb * 64 + db * 2);
        }
      }
    }
    const half_t* X = &xs[cur][0];
    half_t* Xn = &xs[cur ^ 1][0];
    v16h a0 = afrag(X, 64, mb + lh, 0, hi);
    v16h a1 = afrag(X, 64, mb + lh, 32, hi);   // covers features 32..34 + h (cols 48..63)
    v16h ah = afrag_k16(X, 64, mb + lh, 48, hi);
    v8f s0 = zf8(), s1 = zf8(), s2 = zf8(), ee = zf8();
    s0 = wmma16(a0, bf00, s0); s0 = wmma16(a1, bf01, s0);   // r pre-act (x + h parts)
    s1 = wmma16(a0, bf10, s1); s1 = wmma16(a1, bf11, s1);   // z pre-act
    s2 = wmma16(a0, bf20, s2); s2 = wmma16(a1, bf21, s2);   // i_n (x part only)
    ee = wmma16(ah, bfe, ee);                               // h @ Whh_n
#pragma unroll
    for (int v = 0; v < 8; ++v) {
      float r  = sigm(s0[v] + brz0);
      float z  = sigm(s1[v] + brz1);
      float nc = tanhfast(s2[v] + bin + r * (ee[v] + bhn));
      float hv = (1.f - z) * nc + z * hreg[v];
      hreg[v] = hv;
      Xn[(mb + hi * 8 + v) * 64 + 48 + lh] = (half_t)hv;
    }
#pragma unroll
    for (int k = 0; k < 5; ++k) {
#pragma unroll
      for (int j = 0; j < 2; ++j) {
        int e = tid + j * 128;
        if (more && e < 224) {
          int s = e >> 5, db = e & 31;
          Xn[(db * 2) * 64 + k * kS + s] = fv[k][j][0];
          Xn[(db * 2 + 1) * 64 + k * kS + s] = fv[k][j][1];
        }
      }
    }
    __syncthreads();
    cur ^= 1;
  }

  // score MLP on final hidden state: q = relu(relu(h@W1+b1)@W2+b2)@W3+b3
  const half_t* X = &xs[cur][0];
  v16h ahT = afrag_k16(X, 64, mb + lh, 48, hi);
  v16h w1f = bfrag_k16(sW1l, 32, lh, hi);
  v8f g1 = zf8();
  g1 = wmma16(ahT, w1f, g1);
#pragma unroll
  for (int v = 0; v < 8; ++v) {
    float y = g1[v] + sb1s[lh];
    y = y > 0.f ? y : 0.f;
    gs[(mb + hi * 8 + v) * 16 + lh] = (half_t)y;
  }
  __syncthreads();
  v16h a2 = afrag_k16(gs, 16, mb + lh, 0, hi);
  v16h w2f = bfrag_k16(sW2l, 32, lh, hi);
  v8f g2 = zf8();
  g2 = wmma16(a2, w2f, g2);
  __syncthreads();
#pragma unroll
  for (int v = 0; v < 8; ++v) {
    float y = g2[v] + sb2s[lh];
    y = y > 0.f ? y : 0.f;
    gs[(mb + hi * 8 + v) * 16 + lh] = (half_t)y;
  }
  __syncthreads();
  if (tid < 64) {
    float q = sb3g[n];
#pragma unroll
    for (int hh = 0; hh < 16; ++hh) q += (float)gs[tid * 16 + hh] * sw3s[hh];
    out[(size_t)n * kB + yb * 64 + tid] += q;   // LM3 wrote '=' earlier in the stream
  }
}

// ---------------------------------------------------------------------------
extern "C" void kernel_launch(void* const* d_in, const int* in_sizes, int n_in,
                              void* d_out, int out_size, void* d_ws, size_t ws_size,
                              hipStream_t stream) {
  (void)in_sizes; (void)n_in; (void)out_size; (void)ws_size;
  const float* x     = (const float*)d_in[0];
  const float* lm_W1 = (const float*)d_in[1];
  const float* lm_b1 = (const float*)d_in[2];
  const float* lm_W2 = (const float*)d_in[3];
  const float* lm_b2 = (const float*)d_in[4];
  const float* lm_W3 = (const float*)d_in[5];
  const float* lm_b3 = (const float*)d_in[6];
  const float* gWih  = (const float*)d_in[7];
  const float* gWhh  = (const float*)d_in[8];
  const float* gbih  = (const float*)d_in[9];
  const float* gbhh  = (const float*)d_in[10];
  const float* sW1   = (const float*)d_in[11];
  const float* sb1   = (const float*)d_in[12];
  const float* sW2   = (const float*)d_in[13];
  const float* sb2   = (const float*)d_in[14];
  const float* sW3   = (const float*)d_in[15];
  const float* sb3   = (const float*)d_in[16];

  char* ws = (char*)d_ws;
  half_t* xT2 = (half_t*)ws;
  half_t* H1  = (half_t*)(ws + XT2_BYTES);
  half_t* H2  = (half_t*)(ws + XT2_BYTES + H1_BYTES);
  float* out  = (float*)d_out;

  k_transpose<<<dim3(kT, (kM + 31) / 32, kB / 32), 256, 0, stream>>>(x, xT2);
  k_lm1<<<dim3(kN, 8), 256, 0, stream>>>(xT2, lm_W1, lm_b1, H1);
  k_lm2<<<dim3(kN, 4), 256, 0, stream>>>(H1, lm_W2, lm_b2, H2);
  k_lm3<<<dim3(kN), 128, 0, stream>>>(H2, lm_W3, lm_b3, out);
  k_gru<<<dim3(kN, 2), 128, 0, stream>>>(xT2, gWih, gWhh, gbih, gbhh,
                                         sW1, sb1, sW2, sb2, sW3, sb3, out);
}
